// ScaledDotProductAttention_75076028334843
// MI455X (gfx1250) — matrix operand order repaired and checked by reading, not compiled, not run on hardware
//
#include <hip/hip_runtime.h>
#include <stdint.h>

#define SEQ 2048
#define DHEAD 128
#define BATCH 8
#define LDS_STRIDE 2052                 // f32 score row stride (2048+4 pad)
#define PB_STRIDE 1028                  // bf16-pair (dword) row stride (1024+4 pad)
#define NWAVES 8
#define ATT_SCALE 0.08838834764831845f  // 1/sqrt(128)

typedef __attribute__((ext_vector_type(16))) __bf16 v16bf;
typedef __attribute__((ext_vector_type(8)))  float  v8f;
typedef __attribute__((ext_vector_type(4)))  float  f32x4;   // native vec for NT stores

union Frag {
    v16bf    v;
    uint32_t u[8];
    uint4    q[2];
};

// pack two f32 into one dword of bf16 (RNE)
__device__ __forceinline__ uint32_t pack2_bf16(float a, float b) {
#if __has_builtin(__builtin_amdgcn_cvt_pk_bf16_f32)
    typedef __attribute__((ext_vector_type(2))) __bf16 v2bf;
    union { v2bf v; uint32_t u; } cv;
    cv.v = __builtin_amdgcn_cvt_pk_bf16_f32(a, b);
    return cv.u;
#else
    uint32_t ua = __float_as_uint(a);
    uint32_t ub = __float_as_uint(b);
    ua = (ua + 0x7fffu + ((ua >> 16) & 1u)) >> 16;
    ub = (ub + 0x7fffu + ((ub >> 16) & 1u)) & 0xffff0000u;
    return ua | ub;
#endif
}

// ---------------------------------------------------------------------------
// One-time pack: Q,K -> bf16 row-major; V -> bf16 transposed Vt[b][d][s].
// ---------------------------------------------------------------------------
__global__ __launch_bounds__(256)
void prep_pack_kernel(const float* __restrict__ Q, const float* __restrict__ K,
                      const float* __restrict__ V,
                      uint32_t* __restrict__ Qb, uint32_t* __restrict__ Kb,
                      uint32_t* __restrict__ Vt) {
    const uint32_t idx = blockIdx.x * blockDim.x + threadIdx.x;   // dword index
    float2 q = ((const float2*)Q)[idx];
    Qb[idx] = pack2_bf16(q.x, q.y);
    float2 k = ((const float2*)K)[idx];
    Kb[idx] = pack2_bf16(k.x, k.y);
    // Vt: dword = [b][d][spair]; pair = consecutive s for fixed d
    const uint32_t b     = idx >> 17;            // / (D * S/2)
    const uint32_t rem   = idx & 0x1ffffu;
    const uint32_t d     = rem >> 10;            // / (S/2)
    const uint32_t spair = rem & 1023u;
    const float* vp = V + ((size_t)(b * SEQ + 2 * spair)) * DHEAD + d;
    Vt[idx] = pack2_bf16(vp[0], vp[DHEAD]);
}

// ---------------------------------------------------------------------------
// Main attention kernel, pre-packed bf16 operands (fast path).
// One workgroup (8 waves) per (batch, 16 query rows).
// ---------------------------------------------------------------------------
__global__ __launch_bounds__(256)
void sdpa_causal_pk(const uint32_t* __restrict__ Qb, const uint32_t* __restrict__ Kb,
                    const uint32_t* __restrict__ Vt,
                    float* __restrict__ out, float* __restrict__ attn) {
    extern __shared__ char smem_raw[];
    float*    sc     = (float*)smem_raw;                 // 16 x LDS_STRIDE f32 scores/P
    uint32_t* pb     = (uint32_t*)(sc + 16 * LDS_STRIDE);// 16 x PB_STRIDE bf16-pair P
    float*    red    = (float*)(pb + 16 * PB_STRIDE);    // 16x16 reduction scratch
    float*    rowmax = red + 256;
    float*    rowinv = rowmax + 16;

    const int qt    = blockIdx.x;
    const int b     = blockIdx.y;
    const int q0    = qt * 16;
    const int tid   = threadIdx.x;
    const int wave  = tid >> 5;
    const int lane  = tid & 31;
    const int lhalf = lane >> 4;
    const int l16   = lane & 15;
    const int ncols = (qt + 1) * 16;

    // ---- Q A-fragments: direct b128 loads of pre-packed bf16 ----
    Frag qa[4];
    {
        const uint32_t* qrow = Qb + ((size_t)(b * SEQ + q0 + l16)) * (DHEAD / 2);
        #pragma unroll
        for (int c = 0; c < 4; ++c) {
            const int base = c * 16 + 4 * lhalf;       // dword offset: K+0 / K+8
            qa[c].q[0] = *(const uint4*)(qrow + base);
            qa[c].q[1] = *(const uint4*)(qrow + base + 8);
        }
    }

    // ---- scores, sub-diagonal tiles (no mask needed): wave w gets w, w+8,...
    for (int j = wave; j < qt; j += NWAVES) {
        const int k0 = j * 16;
        const uint32_t* krow = Kb + ((size_t)(b * SEQ + k0 + l16)) * (DHEAD / 2);
        if (j + NWAVES < qt)    // prefetch next tile's K row (global_prefetch_b8)
            __builtin_prefetch(Kb + ((size_t)(b * SEQ + k0 + NWAVES * 16 + l16)) * (DHEAD / 2), 0, 1);
        // preload all four B fragments so WMMAs overlap outstanding loads
        Frag bf[4];
        #pragma unroll
        for (int c = 0; c < 4; ++c) {
            const int base = c * 16 + 4 * lhalf;       // dword offset: K+0 / K+8
            bf[c].q[0] = *(const uint4*)(krow + base);
            bf[c].q[1] = *(const uint4*)(krow + base + 8);
        }
        v8f acc = {0.f, 0.f, 0.f, 0.f, 0.f, 0.f, 0.f, 0.f};
        #pragma unroll
        for (int c = 0; c < 4; ++c)
            acc = __builtin_amdgcn_wmma_f32_16x16x32_bf16(false, qa[c].v, false, bf[c].v,
                                                          (short)0, acc, false, false);
        #pragma unroll
        for (int i = 0; i < 8; ++i)
            sc[(i + 8 * lhalf) * LDS_STRIDE + k0 + l16] = acc[i] * ATT_SCALE;
    }

    // ---- diagonal tile (the only one needing the causal mask) ----
    if ((qt & (NWAVES - 1)) == wave) {
        const int k0 = q0;
        const uint32_t* krow = Kb + ((size_t)(b * SEQ + k0 + l16)) * (DHEAD / 2);
        Frag bf[4];
        #pragma unroll
        for (int c = 0; c < 4; ++c) {
            const int base = c * 16 + 4 * lhalf;
            bf[c].q[0] = *(const uint4*)(krow + base);
            bf[c].q[1] = *(const uint4*)(krow + base + 8);
        }
        v8f acc = {0.f, 0.f, 0.f, 0.f, 0.f, 0.f, 0.f, 0.f};
        #pragma unroll
        for (int c = 0; c < 4; ++c)
            acc = __builtin_amdgcn_wmma_f32_16x16x32_bf16(false, qa[c].v, false, bf[c].v,
                                                          (short)0, acc, false, false);
        #pragma unroll
        for (int i = 0; i < 8; ++i) {
            const int M = i + 8 * lhalf;
            float v = acc[i] * ATT_SCALE;
            if (l16 > M) v = -1e30f;                   // k > q within diagonal tile
            sc[M * LDS_STRIDE + k0 + l16] = v;
        }
    }

    if (ncols & 31)   // zero-pad 16 cols so last 32-wide P.V chunk reads zeros
        sc[(tid >> 4) * LDS_STRIDE + ncols + (tid & 15)] = 0.f;
    __syncthreads();

    // ---- softmax over [0, ncols), 16 threads per row ----
    {
        const int r  = tid >> 4;
        const int c0 = tid & 15;
        float* row = sc + r * LDS_STRIDE;
        float pm = -3.4e38f;
        for (int col = c0; col < ncols; col += 16) pm = fmaxf(pm, row[col]);
        red[r * 16 + c0] = pm;
        __syncthreads();
        if (c0 == 0) {
            float m = red[r * 16];
            #pragma unroll
            for (int i = 1; i < 16; ++i) m = fmaxf(m, red[r * 16 + i]);
            rowmax[r] = m;
        }
        __syncthreads();
        const float m = rowmax[r];
        float ps = 0.f;
        for (int col = c0; col < ncols; col += 16) {
            const float e = __expf(row[col] - m);
            row[col] = e;
            ps += e;
        }
        red[r * 16 + c0] = ps;
        __syncthreads();
        if (c0 == 0) {
            float s = 0.f;
            #pragma unroll
            for (int i = 0; i < 16; ++i) s += red[r * 16 + i];
            rowinv[r] = 1.0f / s;
        }
        __syncthreads();
        const float inv = rowinv[r];
        for (int col = c0; col < ncols; col += 16) row[col] *= inv;
    }
    __syncthreads();

    const int nch = (ncols + 31) >> 5;   // 32-wide key chunks for P.V

    // ---- convert normalized P to bf16 pairs in LDS (one pass) ----
    {
        const int r  = tid >> 4;
        const int c0 = tid & 15;
        const float* row = sc + r * LDS_STRIDE;
        uint32_t* prow = pb + r * PB_STRIDE;
        const int ndw = nch * 16;
        for (int dw = c0; dw < ndw; dw += 16)
            prow[dw] = pack2_bf16(row[2 * dw], row[2 * dw + 1]);
    }

    // ---- stream attention weights to HBM: non-temporal (write-once data) ----
    {
        const int r  = tid >> 4;
        const int c0 = tid & 15;
        const f32x4 z = {0.f, 0.f, 0.f, 0.f};
        const float* row = sc + r * LDS_STRIDE;
        f32x4* wout = (f32x4*)(attn + ((size_t)(b * SEQ + q0 + r)) * SEQ);
        for (int c4 = c0; c4 < SEQ / 4; c4 += 16) {
            const int col = c4 * 4;
            const f32x4 val = (col < ncols) ? *(const f32x4*)(row + col) : z;
            __builtin_nontemporal_store(val, wout + c4);
        }
    }
    __syncthreads();

    // ---- O = P @ V : wave w owns output columns [16w, 16w+16) ----
    {
        v8f acc = {0.f, 0.f, 0.f, 0.f, 0.f, 0.f, 0.f, 0.f};
        const int dcol = wave * 16 + l16;
        const uint32_t* vrow = Vt + ((size_t)(b * DHEAD + dcol)) * (SEQ / 2);
        for (int c = 0; c < nch; ++c) {
            const int kb2 = c * 16;                    // chunk base in dwords
            Frag pa, vb;
            const uint32_t* prow = pb + l16 * PB_STRIDE + kb2 + 4 * lhalf;
            pa.q[0] = *(const uint4*)(prow);           // ds_load_b128
            pa.q[1] = *(const uint4*)(prow + 8);
            const uint32_t* vp = vrow + kb2 + 4 * lhalf;
            vb.q[0] = *(const uint4*)(vp);             // global_load_b128
            vb.q[1] = *(const uint4*)(vp + 8);
            acc = __builtin_amdgcn_wmma_f32_16x16x32_bf16(false, pa.v, false, vb.v,
                                                          (short)0, acc, false, false);
        }
        #pragma unroll
        for (int i = 0; i < 8; ++i) {
            const int M = i + 8 * lhalf;
            __builtin_nontemporal_store(acc[i],
                out + ((size_t)(b * SEQ + q0 + M)) * DHEAD + dcol);
        }
    }
}

// ---------------------------------------------------------------------------
// Fallback (no workspace): in-kernel f32->bf16 conversion.
// ---------------------------------------------------------------------------
__global__ __launch_bounds__(256)
void sdpa_causal_fb(const float* __restrict__ Q, const float* __restrict__ K,
                    const float* __restrict__ V,
                    float* __restrict__ out, float* __restrict__ attn) {
    extern __shared__ char smem_raw[];
    float* sc     = (float*)smem_raw;
    float* red    = sc + 16 * LDS_STRIDE;
    float* rowmax = red + 256;
    float* rowinv = rowmax + 16;

    const int qt    = blockIdx.x;
    const int b     = blockIdx.y;
    const int q0    = qt * 16;
    const int tid   = threadIdx.x;
    const int wave  = tid >> 5;
    const int lane  = tid & 31;
    const int lhalf = lane >> 4;
    const int l16   = lane & 15;
    const int ncols = (qt + 1) * 16;

    Frag qa[4];
    {
        const float* qrow = Q + ((size_t)(b * SEQ + q0 + l16)) * DHEAD;
        #pragma unroll
        for (int c = 0; c < 4; ++c) {
            const int base = c * 32 + 8 * lhalf;
            #pragma unroll
            for (int i = 0; i < 4; ++i)
                qa[c].u[i]     = pack2_bf16(qrow[base + 2*i],      qrow[base + 2*i + 1]);
            #pragma unroll
            for (int i = 0; i < 4; ++i)
                qa[c].u[4 + i] = pack2_bf16(qrow[base + 16 + 2*i], qrow[base + 16 + 2*i + 1]);
        }
    }

    for (int j = wave; j <= qt; j += NWAVES) {
        const int k0 = j * 16;
        v8f acc = {0.f, 0.f, 0.f, 0.f, 0.f, 0.f, 0.f, 0.f};
        const float* krow = K + ((size_t)(b * SEQ + k0 + l16)) * DHEAD;
        #pragma unroll
        for (int c = 0; c < 4; ++c) {
            Frag bf;
            const int base = c * 32 + 8 * lhalf;
            #pragma unroll
            for (int jj = 0; jj < 8; ++jj)
                bf.u[jj] = pack2_bf16(krow[base + 2*jj + (jj & 4) * 2], krow[base + 2*jj + (jj & 4) * 2 + 1]);
            acc = __builtin_amdgcn_wmma_f32_16x16x32_bf16(false, qa[c].v, false, bf.v,
                                                          (short)0, acc, false, false);
        }
        #pragma unroll
        for (int i = 0; i < 8; ++i) {
            const int M = i + 8 * lhalf;
            float v = acc[i] * ATT_SCALE;
            if (k0 + l16 > q0 + M) v = -1e30f;
            sc[M * LDS_STRIDE + k0 + l16] = v;
        }
    }

    if (ncols & 31)
        sc[(tid >> 4) * LDS_STRIDE + ncols + (tid & 15)] = 0.f;
    __syncthreads();

    {
        const int r  = tid >> 4;
        const int c0 = tid & 15;
        float* row = sc + r * LDS_STRIDE;
        float pm = -3.4e38f;
        for (int col = c0; col < ncols; col += 16) pm = fmaxf(pm, row[col]);
        red[r * 16 + c0] = pm;
        __syncthreads();
        if (c0 == 0) {
            float m = red[r * 16];
            #pragma unroll
            for (int i = 1; i < 16; ++i) m = fmaxf(m, red[r * 16 + i]);
            rowmax[r] = m;
        }
        __syncthreads();
        const float m = rowmax[r];
        float ps = 0.f;
        for (int col = c0; col < ncols; col += 16) {
            const float e = __expf(row[col] - m);
            row[col] = e;
            ps += e;
        }
        red[r * 16 + c0] = ps;
        __syncthreads();
        if (c0 == 0) {
            float s = 0.f;
            #pragma unroll
            for (int i = 0; i < 16; ++i) s += red[r * 16 + i];
            rowinv[r] = 1.0f / s;
        }
        __syncthreads();
        const float inv = rowinv[r];
        for (int col = c0; col < ncols; col += 16) row[col] *= inv;
    }
    __syncthreads();

    {
        const int r  = tid >> 4;
        const int c0 = tid & 15;
        const f32x4 z = {0.f, 0.f, 0.f, 0.f};
        const float* row = sc + r * LDS_STRIDE;
        f32x4* wout = (f32x4*)(attn + ((size_t)(b * SEQ + q0 + r)) * SEQ);
        for (int c4 = c0; c4 < SEQ / 4; c4 += 16) {
            const int col = c4 * 4;
            const f32x4 val = (col < ncols) ? *(const f32x4*)(row + col) : z;
            __builtin_nontemporal_store(val, wout + c4);
        }
    }

    {
        v8f acc = {0.f, 0.f, 0.f, 0.f, 0.f, 0.f, 0.f, 0.f};
        const int dcol = wave * 16 + l16;
        const int nch  = (ncols + 31) >> 5;
        for (int c = 0; c < nch; ++c) {
            const int kbase = c * 32;
            Frag pa, vb;
            const float* prow = sc + l16 * LDS_STRIDE + kbase + 8 * lhalf;
            #pragma unroll
            for (int i = 0; i < 4; ++i)
                pa.u[i]     = pack2_bf16(prow[2*i],      prow[2*i + 1]);
            #pragma unroll
            for (int i = 0; i < 4; ++i)
                pa.u[4 + i] = pack2_bf16(prow[16 + 2*i], prow[16 + 2*i + 1]);
            const float* vcol = V + ((size_t)(b * SEQ + kbase + 8 * lhalf)) * DHEAD + dcol;
            #pragma unroll
            for (int jj = 0; jj < 8; ++jj)
                vb.u[jj] = pack2_bf16(vcol[(2*jj + (jj & 4) * 2) * DHEAD], vcol[(2*jj + (jj & 4) * 2 + 1) * DHEAD]);
            acc = __builtin_amdgcn_wmma_f32_16x16x32_bf16(false, pa.v, false, vb.v,
                                                          (short)0, acc, false, false);
        }
        #pragma unroll
        for (int i = 0; i < 8; ++i) {
            const int M = i + 8 * lhalf;
            out[((size_t)(b * SEQ + q0 + M)) * DHEAD + dcol] = acc[i];
        }
    }
}

extern "C" void kernel_launch(void* const* d_in, const int* in_sizes, int n_in,
                              void* d_out, int out_size, void* d_ws, size_t ws_size,
                              hipStream_t stream) {
    const float* Q = (const float*)d_in[0];
    const float* K = (const float*)d_in[1];
    const float* V = (const float*)d_in[2];
    // d_in[3] = causal mask; causality applied analytically in-kernel.
    float* out  = (float*)d_out;
    float* attn = out + (size_t)BATCH * SEQ * DHEAD;

    const size_t nElem  = (size_t)BATCH * SEQ * DHEAD;     // 2,097,152
    const size_t needWs = 3 * nElem * sizeof(uint16_t);    // ~12.6 MB bf16 packs
    dim3 grid(SEQ / 16, BATCH);

    if (ws_size >= needWs) {
        uint32_t* Qb = (uint32_t*)d_ws;
        uint32_t* Kb = Qb + nElem / 2;
        uint32_t* Vt = Kb + nElem / 2;
        const int nDw = (int)(nElem / 2);                  // 1,048,576 dwords
        prep_pack_kernel<<<nDw / 256, 256, 0, stream>>>(Q, K, V, Qb, Kb, Vt);
        const size_t lds = (size_t)(16 * LDS_STRIDE) * 4 + (size_t)(16 * PB_STRIDE) * 4
                         + (256 + 16 + 16) * 4;
        sdpa_causal_pk<<<grid, 256, lds, stream>>>(Qb, Kb, Vt, out, attn);
    } else {
        const size_t lds = (size_t)(16 * LDS_STRIDE + 256 + 16 + 16) * sizeof(float);
        sdpa_causal_fb<<<grid, 256, lds, stream>>>(Q, K, V, out, attn);
    }
}
